// GraphModule_78889959293634
// MI455X (gfx1250) — matrix operand order repaired and checked by reading, not compiled, not run on hardware
//
#include <hip/hip_runtime.h>
#include <hip/hip_bf16.h>

#define N_NODES 10000
#define E_EDGES 320000
#define F_IN    256
#define F_PROJ  200
#define F_PAD   208   // 13 * 16

typedef __attribute__((ext_vector_type(16))) __bf16 v16bf;
typedef __attribute__((ext_vector_type(8)))  float  v8f;

// ---------------------------------------------------------------- init
__global__ void init_kernel(unsigned* gmax_bits, float* gsum, int* flags) {
    int tid = blockIdx.x * blockDim.x + threadIdx.x;
    if (tid == 0) { *gmax_bits = 0u; *gsum = 1.0f; }
    for (int i = tid; i < N_NODES; i += gridDim.x * blockDim.x) flags[i] = 0;
}

// ---------------------------------------------------------------- edge_attr copy
__global__ void copy_kernel(const float4* __restrict__ in, float4* __restrict__ out,
                            long long n4) {
    long long i = (long long)blockIdx.x * blockDim.x + threadIdx.x;
    long long stride = (long long)gridDim.x * blockDim.x;
    for (; i < n4; i += stride) out[i] = in[i];
}

// ---------------------------------------------------------------- alpha via WMMA
// One wave = 16 edges. diff = x[dst]-x[src] (bf16), GEMM vs W_proj (bf16 in LDS,
// transposed), then alpha_e = relu( sum_n |t[e][n]| * W_att[n] + b_att ).
__global__ __launch_bounds__(256, 1)
void alpha_wmma_kernel(const float* __restrict__ x,
                       const long long* __restrict__ src_idx,
                       const long long* __restrict__ dst_idx,
                       const float* __restrict__ W_proj,
                       const float* __restrict__ W_att,
                       const float* __restrict__ b_att,
                       float* __restrict__ alpha,
                       unsigned* __restrict__ gmax_bits)
{
    extern __shared__ char smem_raw[];
    __bf16* Wl = (__bf16*)smem_raw;                       // [F_PAD][F_IN] transposed
    float*  Wa = (float*)(smem_raw + (size_t)F_PAD * F_IN * sizeof(__bf16));

    const int tid = threadIdx.x;   // 256 threads = 8 waves

    // stage W_proj^T as bf16 (pad cols 200..207 with zeros), W_att padded too
    for (int i = tid; i < F_PAD * F_IN; i += 256) {
        int n = i >> 8;            // output column
        int k = i & 255;           // input feature
        float v = (n < F_PROJ) ? W_proj[k * F_PROJ + n] : 0.0f;
        Wl[i] = (__bf16)v;
    }
    for (int i = tid; i < F_PAD; i += 256)
        Wa[i] = (i < F_PROJ) ? W_att[i] : 0.0f;
    __syncthreads();

    const int wave = tid >> 5;
    const int lane = tid & 31;
    const int half = lane >> 4;    // 0: lanes 0-15, 1: lanes 16-31
    const int row  = lane & 15;

    const long long tile = (long long)blockIdx.x * 8 + wave;   // 20000 tiles
    const long long e0   = tile * 16;
    const long long e    = e0 + row;

    const long long s = src_idx[e];
    const long long d = dst_idx[e];
    const float4* xs = (const float4*)(x + s * F_IN);
    const float4* xd = (const float4*)(x + d * F_IN);

    // Build 8 A-fragments (16x32 bf16 each) covering K=0..255.
    // A 16x32 layout (matches B): lane half h, element j -> K = 16h + j
    v16bf A[8];
    #pragma unroll
    for (int kc = 0; kc < 8; ++kc) {
        const int k0 = kc * 32 + half * 16;       // j = 0..7
        const int k1 = kc * 32 + half * 16 + 8;   // j = 8..15
        float4 d0 = xd[k0 >> 2], s0 = xs[k0 >> 2];
        float4 d1 = xd[(k0 >> 2) + 1], s1 = xs[(k0 >> 2) + 1];
        float4 d2 = xd[k1 >> 2], s2 = xs[k1 >> 2];
        float4 d3 = xd[(k1 >> 2) + 1], s3 = xs[(k1 >> 2) + 1];
        float dv[16];
        dv[0]  = d0.x - s0.x; dv[1]  = d0.y - s0.y; dv[2]  = d0.z - s0.z; dv[3]  = d0.w - s0.w;
        dv[4]  = d1.x - s1.x; dv[5]  = d1.y - s1.y; dv[6]  = d1.z - s1.z; dv[7]  = d1.w - s1.w;
        dv[8]  = d2.x - s2.x; dv[9]  = d2.y - s2.y; dv[10] = d2.z - s2.z; dv[11] = d2.w - s2.w;
        dv[12] = d3.x - s3.x; dv[13] = d3.y - s3.y; dv[14] = d3.z - s3.z; dv[15] = d3.w - s3.w;
        #pragma unroll
        for (int j = 0; j < 16; ++j) A[kc][j] = (__bf16)dv[j];
    }

    const float batt = b_att[0];
    float partial[8];
    #pragma unroll
    for (int r = 0; r < 8; ++r) partial[r] = 0.0f;

    const int col = row;   // this lane's N-column within the tile
    for (int nt = 0; nt < 13; ++nt) {
        v8f acc = {};
        #pragma unroll
        for (int kc = 0; kc < 8; ++kc) {
            // B 32x16 bf16 layout: lane half h holds K = 16h + j for column (lane&15)
            const v16bf B = *(const v16bf*)(Wl + (nt * 16 + col) * F_IN + kc * 32 + half * 16);
            acc = __builtin_amdgcn_wmma_f32_16x16x32_bf16(
                      false, A[kc], false, B, (short)0, acc, false, false);
        }
        const float wa = Wa[nt * 16 + col];
        #pragma unroll
        for (int r = 0; r < 8; ++r) partial[r] += fabsf(acc[r]) * wa;
    }

    // Reduce each partial across the 16 lanes of this half (sum over columns)
    #pragma unroll
    for (int r = 0; r < 8; ++r) {
        float v = partial[r];
        v += __shfl_xor(v, 1, 32);
        v += __shfl_xor(v, 2, 32);
        v += __shfl_xor(v, 4, 32);
        v += __shfl_xor(v, 8, 32);
        partial[r] = v;
    }

    if (row == 0) {
        float lmax = 0.0f;
        #pragma unroll
        for (int r = 0; r < 8; ++r) {
            const long long m = e0 + r + 8 * half;   // C layout: M = r + 8*half
            float a = partial[r] + batt;
            a = a > 0.0f ? a : 0.0f;                 // relu
            alpha[m] = a;
            lmax = a > lmax ? a : lmax;
        }
        atomicMax(gmax_bits, __float_as_uint(lmax)); // valid: relu => bits monotone
    }
}

// ---------------------------------------------------------------- exp + partial sums + flags
__global__ void exp_kernel(float* __restrict__ alpha,
                           const unsigned* __restrict__ gmax_bits,
                           const long long* __restrict__ dst_idx,
                           int* __restrict__ flags,
                           float* __restrict__ partials) {
    __shared__ float red[256];
    const int tid = blockIdx.x * 256 + threadIdx.x;
    const float gmax = __uint_as_float(*gmax_bits);
    float ev = 0.0f;
    if (tid < E_EDGES) {
        float a = alpha[tid];
        ev = __expf(a - gmax);
        alpha[tid] = ev;
        flags[dst_idx[tid]] = 1;
    }
    red[threadIdx.x] = ev;
    __syncthreads();
    for (int st = 128; st > 0; st >>= 1) {
        if (threadIdx.x < st) red[threadIdx.x] += red[threadIdx.x + st];
        __syncthreads();
    }
    if (threadIdx.x == 0) partials[blockIdx.x] = red[0];
}

__global__ void reduce_kernel(const float* __restrict__ partials, int n,
                              float* __restrict__ gsum) {
    __shared__ float red[256];
    float s = 0.0f;
    for (int i = threadIdx.x; i < n; i += 256) s += partials[i];
    red[threadIdx.x] = s;
    __syncthreads();
    for (int st = 128; st > 0; st >>= 1) {
        if (threadIdx.x < st) red[threadIdx.x] += red[threadIdx.x + st];
        __syncthreads();
    }
    if (threadIdx.x == 0) *gsum = red[0];
}

// ---------------------------------------------------------------- normalize + scatters
__global__ void scatter1_kernel(float* __restrict__ alpha,
                                const float* __restrict__ gsum,
                                const long long* __restrict__ src_idx,
                                const long long* __restrict__ dst_idx,
                                float* __restrict__ edge_out) {
    const int tid = blockIdx.x * 256 + threadIdx.x;
    if (tid < E_EDGES) {
        const float a = alpha[tid] / *gsum;
        alpha[tid] = a;
        edge_out[src_idx[tid] * (long long)N_NODES + dst_idx[tid]] = a;
    }
}

__global__ void scatter2_kernel(const float* __restrict__ alpha,
                                const long long* __restrict__ src_idx,
                                const long long* __restrict__ dst_idx,
                                float* __restrict__ edge_out) {
    const int tid = blockIdx.x * 256 + threadIdx.x;
    if (tid < E_EDGES)
        edge_out[dst_idx[tid] * (long long)N_NODES + src_idx[tid]] = alpha[tid];
}

// ---------------------------------------------------------------- agg = flag ? x : 0
__global__ void agg_kernel(const float* __restrict__ x,
                           const int* __restrict__ flags,
                           float* __restrict__ out) {
    const int tid = blockIdx.x * 256 + threadIdx.x;
    if (tid < N_NODES * F_IN) {
        const int n = tid >> 8;
        out[tid] = flags[n] ? x[tid] : 0.0f;
    }
}

// ---------------------------------------------------------------- launcher
extern "C" void kernel_launch(void* const* d_in, const int* in_sizes, int n_in,
                              void* d_out, int out_size, void* d_ws, size_t ws_size,
                              hipStream_t stream) {
    const float*     x         = (const float*)d_in[0];
    const float*     edge_attr = (const float*)d_in[1];
    const long long* eidx      = (const long long*)d_in[2];
    const float*     W_proj    = (const float*)d_in[3];
    // d_in[4] = b_proj: cancels in p_i - p_j, unused
    const float*     W_att     = (const float*)d_in[5];
    const float*     b_att     = (const float*)d_in[6];

    const long long* src = eidx;
    const long long* dst = eidx + E_EDGES;

    float* out_agg   = (float*)d_out;
    float* out_edges = out_agg + (long long)N_NODES * F_IN;

    char*     ws        = (char*)d_ws;
    unsigned* gmax_bits = (unsigned*)ws;
    float*    gsum      = (float*)(ws + 4);
    int*      flags     = (int*)(ws + 128);
    float*    alpha     = (float*)(ws + 40192);                          // 128 + 40000 rounded
    float*    partials  = (float*)(ws + 40192 + (size_t)E_EDGES * 4);

    init_kernel<<<40, 256, 0, stream>>>(gmax_bits, gsum, flags);

    const long long n4 = (long long)N_NODES * N_NODES / 4;
    copy_kernel<<<4096, 256, 0, stream>>>((const float4*)edge_attr, (float4*)out_edges, n4);

    const size_t smem = (size_t)F_PAD * F_IN * sizeof(__bf16) + F_PAD * sizeof(float); // ~107 KB
    alpha_wmma_kernel<<<E_EDGES / 128, 256, smem, stream>>>(
        x, src, dst, W_proj, W_att, b_att, alpha, gmax_bits);

    exp_kernel<<<E_EDGES / 256, 256, 0, stream>>>(alpha, gmax_bits, dst, flags, partials);
    reduce_kernel<<<1, 256, 0, stream>>>(partials, E_EDGES / 256, gsum);

    scatter1_kernel<<<E_EDGES / 256, 256, 0, stream>>>(alpha, gsum, src, dst, out_edges);
    scatter2_kernel<<<E_EDGES / 256, 256, 0, stream>>>(alpha, src, dst, out_edges);

    agg_kernel<<<(N_NODES * F_IN) / 256, 256, 0, stream>>>(x, flags, out_agg);
}
